// TriangleMultiplication_26637387170082
// MI455X (gfx1250) — matrix operand order repaired and checked by reading, not compiled, not run on hardware
//
#include <hip/hip_runtime.h>

// ---------------------------------------------------------------------------
// Triangle multiplicative update (outgoing), B=1, L=512, D=128, DI=64.
//
// Dominant term: per-channel LxL contraction (17.2 GFLOP) -> bf16 WMMA
// (v_wmma_f32_16x16x32_bf16). bf16 intermediates g1/g2 (32 MB each) fit in
// the 192 MB L2, so kernel-2 tile re-reads are L2 hits. Memory floor ~0.5 GB
// => ~22 us at 23.3 TB/s. All f32->bf16 conversion uses native casts (single
// v_cvt, often packed) instead of integer RNE emulation; p is parked in LDS
// as bf16 so the Wp-projection WMMA consumes it with raw ds_load_b128.
// ---------------------------------------------------------------------------

#define LL 512
#define DD 128
#define DI 64

typedef __attribute__((ext_vector_type(16))) __bf16 v16bf;
typedef __attribute__((ext_vector_type(8)))  float  v8f;

union BFrag {
  v16bf v;
  __bf16 b[16];
  uint4 q[2];
};

union Pack2 { __bf16 b[2]; unsigned int u; };
union Pack8 { __bf16 b[8]; uint4 q; };

// ---------------------------------------------------------------------------
// Kernel 1: g = bf16(pair @ W + b), written channel-major in WMMA-friendly
// layout. mode 0: row-slabs  -> g1ik[d][i][k]   (A-operand, k contiguous)
//          mode 1: col-slabs -> g2jk[d][j][k]   (B^T,      k contiguous)
// Each wave owns one 16-row slab; W is staged transposed+bf16 in LDS.
// ---------------------------------------------------------------------------
__global__ __launch_bounds__(256) void proj_bf16_kernel(
    const float* __restrict__ pair, const float* __restrict__ W,
    const float* __restrict__ bias, unsigned short* __restrict__ out, int mode)
{
  __shared__ __align__(16) __bf16 WtT[64][136];  // [n][d], padded

  const int tid = threadIdx.x;
  for (int idx = tid; idx < 128 * 64; idx += 256) {
    int n = idx & 63, d = idx >> 6;
    WtT[n][d] = (__bf16)W[d * 64 + n];
  }
  __syncthreads();

  const int wave = tid >> 5, lane = tid & 31;
  const int lrow = lane & 15, half = lane >> 4;
  const int slab = blockIdx.x * 8 + wave;          // 16384 slabs total

  long rowBase, rowStride;
  int outRow, colBase;
  if (mode == 0) {                                  // g1: i fixed, k contiguous
    int r0 = slab << 4;
    rowBase = (long)r0 * DD; rowStride = DD;
    outRow = r0 >> 9; colBase = r0 & (LL - 1);
  } else {                                          // g2: j fixed, k contiguous
    int j = slab & (LL - 1);
    int q0 = (slab >> 9) << 4;
    rowBase = ((long)q0 * LL + j) * DD; rowStride = (long)LL * DD;
    outRow = j; colBase = q0;
  }

  const float* rp = pair + rowBase + (long)lrow * rowStride;

  v8f acc[4];
  #pragma unroll
  for (int nt = 0; nt < 4; ++nt) acc[nt] = v8f{};

  #pragma unroll
  for (int ks = 0; ks < 4; ++ks) {
    const int k0 = ks * 32;
    const int s1 = k0 + 8 * half;
    float4 f0 = *(const float4*)(rp + s1);
    float4 f1 = *(const float4*)(rp + s1 + 4);
    float4 f2 = *(const float4*)(rp + s1 + 16);
    float4 f3 = *(const float4*)(rp + s1 + 20);
    BFrag a;
    a.b[0]  = (__bf16)f0.x; a.b[1]  = (__bf16)f0.y;
    a.b[2]  = (__bf16)f0.z; a.b[3]  = (__bf16)f0.w;
    a.b[4]  = (__bf16)f1.x; a.b[5]  = (__bf16)f1.y;
    a.b[6]  = (__bf16)f1.z; a.b[7]  = (__bf16)f1.w;
    a.b[8]  = (__bf16)f2.x; a.b[9]  = (__bf16)f2.y;
    a.b[10] = (__bf16)f2.z; a.b[11] = (__bf16)f2.w;
    a.b[12] = (__bf16)f3.x; a.b[13] = (__bf16)f3.y;
    a.b[14] = (__bf16)f3.z; a.b[15] = (__bf16)f3.w;

    #pragma unroll
    for (int nt = 0; nt < 4; ++nt) {
      BFrag bfr;
      const __bf16* wp = &WtT[nt * 16 + lrow][k0 + 8 * half];
      bfr.q[0] = *(const uint4*)wp;
      bfr.q[1] = *(const uint4*)(wp + 16);
      acc[nt] = __builtin_amdgcn_wmma_f32_16x16x32_bf16(
          false, a.v, false, bfr.v, (short)0, acc[nt], false, false);
    }
  }

  #pragma unroll
  for (int nt = 0; nt < 4; ++nt) {
    const int ch = nt * 16 + lrow;
    const float bv = bias[ch];
    Pack8 pk;
    #pragma unroll
    for (int r = 0; r < 8; ++r) pk.b[r] = (__bf16)(acc[nt][r] + bv);
    *(uint4*)(out + ((size_t)ch * LL + outRow) * LL + colBase + 8 * half) = pk.q;
  }
}

// ---------------------------------------------------------------------------
// Kernel 2: fused triangle contraction + mask + Wp projection + bias +
// residual + LayerNorm for one 16x16 (i,j) tile. 8 waves x 8 channels each.
// ---------------------------------------------------------------------------
#define PB_STRIDE  72    // bf16 p tile: 64 ch + 8 pad halves (144B rows, 16B-aligned)
#define WPT_STRIDE 72    // bf16 WpT:    64 ch + 8 pad halves
#define Y_STRIDE   132   // f32 y:       128 dd + 4 pad floats

__global__ __launch_bounds__(256) void tri_fused_kernel(
    const unsigned short* __restrict__ g1, const unsigned short* __restrict__ g2,
    const float* __restrict__ mask, const float* __restrict__ Wp,
    const float* __restrict__ bp, const float* __restrict__ pair,
    const float* __restrict__ gamma, const float* __restrict__ beta,
    float* __restrict__ out)
{
  extern __shared__ __align__(16) char smem_raw[];
  __bf16* p_bf  = (__bf16*)smem_raw;                       // 256*72 bf16 (36 KB)
  __bf16* WpT   = p_bf + 256 * PB_STRIDE;                  // 128*72 bf16 (18 KB)
  float*  y_lds = (float*)(WpT + 128 * WPT_STRIDE);        // 256*132 f32 (132 KB)
  float*  m_lds = y_lds + 256 * Y_STRIDE;                  // 256 f32

  const int tid = threadIdx.x;
  const int i0 = blockIdx.y * 16, j0 = blockIdx.x * 16;

  // stage mask tile + transposed bf16 Wp:  WpT[dd][d] = bf16(Wp[d][dd])
  m_lds[tid] = mask[(size_t)(i0 + (tid >> 4)) * LL + (j0 + (tid & 15))];
  for (int idx = tid; idx < 64 * 128; idx += 256) {
    int dd = idx & 127, d = idx >> 7;
    WpT[dd * WPT_STRIDE + d] = (__bf16)Wp[d * 128 + dd];
  }
  __syncthreads();

  const int wave = tid >> 5, lane = tid & 31;
  const int lrow = lane & 15, half = lane >> 4;

  // ---- phase 1: p[i,j,d] = sum_k g1[i,k,d] g2[k,j,d], channels wave*8..+7 --
  v8f acc[8];
  #pragma unroll
  for (int c = 0; c < 8; ++c) acc[c] = v8f{};

  for (int k0 = 0; k0 < LL; k0 += 32) {
    #pragma unroll
    for (int c = 0; c < 8; ++c) {
      const int d = wave * 8 + c;
      BFrag a, b;
      const unsigned short* ap =
          g1 + ((size_t)d * LL + i0 + lrow) * LL + k0 + 8 * half;
      a.q[0] = *(const uint4*)ap;
      a.q[1] = *(const uint4*)(ap + 16);
      const unsigned short* bpt =
          g2 + ((size_t)d * LL + j0 + lrow) * LL + k0 + 8 * half;
      b.q[0] = *(const uint4*)bpt;
      b.q[1] = *(const uint4*)(bpt + 16);
      acc[c] = __builtin_amdgcn_wmma_f32_16x16x32_bf16(
          false, a.v, false, b.v, (short)0, acc[c], false, false);
    }
  }

  // mask, convert to bf16, park p in LDS as [pos][ch] (channel pairs packed)
  float mv[8];
  #pragma unroll
  for (int r = 0; r < 8; ++r) mv[r] = m_lds[(r + 8 * half) * 16 + lrow];
  #pragma unroll
  for (int r = 0; r < 8; ++r) {
    const int pos = (r + 8 * half) * 16 + lrow;
    #pragma unroll
    for (int cp = 0; cp < 4; ++cp) {
      Pack2 pk;
      pk.b[0] = (__bf16)(acc[2 * cp][r] * mv[r]);
      pk.b[1] = (__bf16)(acc[2 * cp + 1][r] * mv[r]);
      *(unsigned int*)(p_bf + pos * PB_STRIDE + wave * 8 + 2 * cp) = pk.u;
    }
  }
  __syncthreads();

  // ---- phase 2: y = p @ Wp  (positions x 64ch @ 64x128), WMMA again --------
  #pragma unroll
  for (int rt = 0; rt < 2; ++rt) {
    const int pos = wave * 32 + rt * 16 + lrow;
    BFrag a[2];
    #pragma unroll
    for (int ksx = 0; ksx < 2; ++ksx) {
      const __bf16* px = p_bf + pos * PB_STRIDE + ksx * 32 + 8 * half;
      a[ksx].q[0] = *(const uint4*)px;
      a[ksx].q[1] = *(const uint4*)(px + 16);
    }
    #pragma unroll
    for (int nt = 0; nt < 8; ++nt) {
      v8f pacc = v8f{};
      #pragma unroll
      for (int ksx = 0; ksx < 2; ++ksx) {
        BFrag b;
        const __bf16* wpp =
            WpT + (nt * 16 + lrow) * WPT_STRIDE + ksx * 32 + 8 * half;
        b.q[0] = *(const uint4*)wpp;
        b.q[1] = *(const uint4*)(wpp + 16);
        pacc = __builtin_amdgcn_wmma_f32_16x16x32_bf16(
            false, a[ksx].v, false, b.v, (short)0, pacc, false, false);
      }
      #pragma unroll
      for (int r = 0; r < 8; ++r) {
        const int pp = wave * 32 + rt * 16 + r + 8 * half;
        y_lds[pp * Y_STRIDE + nt * 16 + lrow] = pacc[r];
      }
    }
  }
  __syncthreads();

  // ---- phase 3: x = pair + (y + bp); LayerNorm over D=128 per position -----
  {
    const int m = tid >> 4, n = tid & 15;
    const float* prow = pair + ((size_t)(i0 + m) * LL + (j0 + n)) * DD;
    const float* yrow = y_lds + tid * Y_STRIDE;
    float4 xv[32];
    float s = 0.f, s2 = 0.f;
    #pragma unroll
    for (int q = 0; q < 32; ++q) {
      float4 yv = *(const float4*)(yrow + q * 4);
      float4 pv = *(const float4*)(prow + q * 4);
      float4 bv = *(const float4*)(bp + q * 4);
      float4 x;
      x.x = yv.x + bv.x + pv.x;
      x.y = yv.y + bv.y + pv.y;
      x.z = yv.z + bv.z + pv.z;
      x.w = yv.w + bv.w + pv.w;
      xv[q] = x;
      s  += x.x + x.y + x.z + x.w;
      s2 += x.x * x.x + x.y * x.y + x.z * x.z + x.w * x.w;
    }
    const float mu  = s * (1.f / 128.f);
    const float var = s2 * (1.f / 128.f) - mu * mu;
    const float rs  = rsqrtf(var + 1e-5f);
    float* orow = out + ((size_t)(i0 + m) * LL + (j0 + n)) * DD;
    #pragma unroll
    for (int q = 0; q < 32; ++q) {
      float4 g  = *(const float4*)(gamma + q * 4);
      float4 bb = *(const float4*)(beta + q * 4);
      float4 o;
      o.x = (xv[q].x - mu) * rs * g.x + bb.x;
      o.y = (xv[q].y - mu) * rs * g.y + bb.y;
      o.z = (xv[q].z - mu) * rs * g.z + bb.z;
      o.w = (xv[q].w - mu) * rs * g.w + bb.w;
      *(float4*)(orow + q * 4) = o;
    }
  }
}

// ---------------------------------------------------------------------------
extern "C" void kernel_launch(void* const* d_in, const int* in_sizes, int n_in,
                              void* d_out, int out_size, void* d_ws, size_t ws_size,
                              hipStream_t stream) {
  const float* pair  = (const float*)d_in[0];
  const float* mask  = (const float*)d_in[1];
  const float* Wg1   = (const float*)d_in[2];
  const float* bg1   = (const float*)d_in[3];
  const float* Wg2   = (const float*)d_in[4];
  const float* bg2   = (const float*)d_in[5];
  const float* Wp    = (const float*)d_in[6];
  const float* bp    = (const float*)d_in[7];
  const float* gamma = (const float*)d_in[8];
  const float* beta  = (const float*)d_in[9];
  float* out = (float*)d_out;

  // workspace: g1ik bf16 [64][512][512] then g2jk bf16 [64][512][512] (64 MB)
  unsigned short* g1 = (unsigned short*)d_ws;
  unsigned short* g2 = g1 + (size_t)DI * LL * LL;

  proj_bf16_kernel<<<2048, 256, 0, stream>>>(pair, Wg1, bg1, g1, 0);
  proj_bf16_kernel<<<2048, 256, 0, stream>>>(pair, Wg2, bg2, g2, 1);

  const size_t smem = (size_t)(256 * PB_STRIDE) * 2 +
                      (size_t)(128 * WPT_STRIDE) * 2 +
                      (size_t)(256 * Y_STRIDE) * 4 + 256 * 4;
  tri_fused_kernel<<<dim3(32, 32), 256, smem, stream>>>(
      g1, g2, mask, Wp, bp, pair, gamma, beta, out);
}
